// GAT_27444841021512
// MI455X (gfx1250) — matrix operand order repaired and checked by reading, not compiled, not run on hardware
//
#include <hip/hip_runtime.h>
#include <hip/hip_bf16.h>

// ---------------------------------------------------------------------------
// GAT (3x GATConv + BN/ReLU + MLP classifier) for MI455X (gfx1250, wave32).
// Dense feature GEMMs run on v_wmma_f32_16x16x32_bf16 (f32 accumulate);
// W tiles are staged into LDS with global_load_async_to_lds_b128 (ASYNCcnt);
// edge softmax/scatter phase stays fp32 (it is HBM/L2-bandwidth bound).
// ---------------------------------------------------------------------------

typedef __attribute__((ext_vector_type(16))) __bf16 v16bf;
typedef __attribute__((ext_vector_type(8)))  float  v8f;

#define N_NODES 50000
#define N_EDGES 200000
#define F_IN    256
#define CDIM    128
#define HEADS   6
#define FDIM    768          // HEADS * CDIM
#define NCLS    40
#define NEG_SLOPE 0.2f

// order-preserving float<->uint encoding for atomic max on floats
static __device__ __forceinline__ unsigned f2ord(float f) {
  unsigned u = __float_as_uint(f);
  return (u & 0x80000000u) ? ~u : (u | 0x80000000u);
}
static __device__ __forceinline__ float ord2f(unsigned u) {
  return (u & 0x80000000u) ? __uint_as_float(u ^ 0x80000000u)
                           : __uint_as_float(~u);
}

// ---------------------------------------------------------------------------
// fp32 -> bf16 cast (grid-stride)
// ---------------------------------------------------------------------------
__global__ void k_cast_bf16(const float* __restrict__ src,
                            __bf16* __restrict__ dst, long n) {
  long i = (long)blockIdx.x * blockDim.x + threadIdx.x;
  long stride = (long)gridDim.x * blockDim.x;
  for (; i < n; i += stride) dst[i] = (__bf16)src[i];
}

// W [K][Nout] fp32 -> Wt [Nout][K] bf16 (so LDS fill + B fragments are contiguous)
__global__ void k_wtrans(const float* __restrict__ W, __bf16* __restrict__ Wt,
                         int K, int Nout) {
  int t = blockIdx.x * blockDim.x + threadIdx.x;
  if (t >= K * Nout) return;
  int c = t % Nout, k = t / Nout;
  Wt[(size_t)c * K + k] = (__bf16)W[t];
}

// ---------------------------------------------------------------------------
// bf16 WMMA GEMM: C[M][Nout] = A[M][K] * W  (Wt given transposed [Nout][K])
// workgroup = 8 waves; wave -> 16 rows x 64 cols; W strip staged in LDS via
// async-to-LDS copies. K multiple of 256, Nout multiple of 64, M mult of 16.
// ---------------------------------------------------------------------------
#define KC 256
#define LDS_STRIDE 272   // KC + 16 pad (bank spread for b128 reads)

__global__ __launch_bounds__(256) void k_gemm_bf16(
    const __bf16* __restrict__ A, const __bf16* __restrict__ Wt,
    float* __restrict__ C, int M, int K, int Nout) {
  __shared__ __bf16 ldsW[64 * LDS_STRIDE];

  const int tid  = threadIdx.x;
  const int lane = tid & 31;
  const int wid  = tid >> 5;
  const int c0   = blockIdx.y * 64;
  const int r0   = (blockIdx.x * 8 + wid) * 16;
  const bool active = (r0 < M);
  const int arow  = active ? (r0 + (lane & 15)) : 0;
  const int akoff = (lane >> 4) * 8;    // A frag: lane half -> K 0..7 / 8..15 (+16 upper)
  const int bkoff = (lane >> 4) * 8;    // B frag: lane half -> K 0..7 / 8..15 (+16 upper)

  // cooperative LDS fill mapping: thread -> (column, 64-element K segment)
  const int colL = tid >> 2;
  const int segL = (tid & 3) * 64;
  const __bf16* wsrc = Wt + (size_t)(c0 + colL) * K + segL;
  // Destination inside LDS. The generic (flat) shared-aperture address keeps
  // the DS byte offset in its low 32 bits, which is exactly the VDST operand
  // of global_load_async_to_lds_*. Keeping the ptrtoint of the real pointer
  // (instead of a folded pointer difference) also *captures* ldsW, so the
  // asm "memory" clobber is known to alias it and the later ds_loads survive.
  __bf16* wdst = ldsW + colL * LDS_STRIDE + segL;
  const unsigned ldsoff = (unsigned)(size_t)wdst;

  v8f acc[4] = {};
  union Frag { uint4 u[2]; v16bf v; };

  for (int kc = 0; kc < K; kc += KC) {
    __syncthreads();   // previous chunk's LDS reads complete
    {
      // 128 B per thread: 8x async b128 copies, ASYNCcnt-tracked, no VGPR data
      unsigned long long gaddr = (unsigned long long)(wsrc + kc);
      asm volatile(
          "global_load_async_to_lds_b128 %0, %1, off\n\t"
          "global_load_async_to_lds_b128 %0, %1, off offset:16\n\t"
          "global_load_async_to_lds_b128 %0, %1, off offset:32\n\t"
          "global_load_async_to_lds_b128 %0, %1, off offset:48\n\t"
          "global_load_async_to_lds_b128 %0, %1, off offset:64\n\t"
          "global_load_async_to_lds_b128 %0, %1, off offset:80\n\t"
          "global_load_async_to_lds_b128 %0, %1, off offset:96\n\t"
          "global_load_async_to_lds_b128 %0, %1, off offset:112\n\t"
          :
          : "v"(ldsoff), "v"(gaddr), "v"(wdst)  // wdst operand: escape hatch
          : "memory");
      asm volatile("s_wait_asynccnt 0" ::: "memory");
    }
    __syncthreads();   // whole W chunk visible to all waves

    const __bf16* aptr = A + (size_t)arow * K + kc + akoff;
    if (kc + KC < K) __builtin_prefetch(aptr + KC, 0, 0);  // next A chunk -> L2

    for (int kk = 0; kk < KC; kk += 32) {
      Frag fa;
      const uint4* ap = (const uint4*)(aptr + kk);
      fa.u[0] = ap[0];          // K akoff..akoff+7
      fa.u[1] = ap[2];          // K akoff+16..akoff+23
#pragma unroll
      for (int s = 0; s < 4; ++s) {
        Frag fb;
        const uint4* bp =
            (const uint4*)(ldsW + (s * 16 + (lane & 15)) * LDS_STRIDE + kk + bkoff);
        fb.u[0] = bp[0];        // K bkoff..bkoff+7
        fb.u[1] = bp[2];        // K bkoff+16..bkoff+23
        acc[s] = __builtin_amdgcn_wmma_f32_16x16x32_bf16(
            false, fa.v, false, fb.v, (short)0, acc[s], false, false);
      }
    }
  }

  if (active) {
    const int rbase = r0 + ((lane >> 4) << 3);
    const int cbase = c0 + (lane & 15);
#pragma unroll
    for (int s = 0; s < 4; ++s)
#pragma unroll
      for (int i = 0; i < 8; ++i)
        C[(size_t)(rbase + i) * Nout + cbase + s * 16] = acc[s][i];
  }
}

// ---------------------------------------------------------------------------
// per-node per-head attention projections alpha_s/alpha_d (dot over C=128)
// ---------------------------------------------------------------------------
__global__ void k_alpha(const float* __restrict__ h, const float* __restrict__ a_s,
                        const float* __restrict__ a_d, float* __restrict__ as_o,
                        float* __restrict__ ad_o, int H_, int F_) {
  int t = blockIdx.x * blockDim.x + threadIdx.x;
  if (t >= N_NODES * H_) return;
  int n = t / H_, hd = t % H_;
  const float4* hp = (const float4*)(h + (size_t)n * F_ + hd * CDIM);
  const float4* sp = (const float4*)(a_s + hd * CDIM);
  const float4* dp = (const float4*)(a_d + hd * CDIM);
  float accs = 0.f, accd = 0.f;
#pragma unroll 8
  for (int i = 0; i < CDIM / 4; ++i) {
    float4 hv = hp[i], sv = sp[i], dv = dp[i];
    accs += hv.x * sv.x + hv.y * sv.y + hv.z * sv.z + hv.w * sv.w;
    accd += hv.x * dv.x + hv.y * dv.y + hv.z * dv.z + hv.w * dv.w;
  }
  as_o[t] = accs;
  ad_o[t] = accd;
}

// ---------------------------------------------------------------------------
// edge pass 1: segment max of leaky_relu(alpha_s[src]+alpha_d[dst])
// virtual edges [E, E+N) are the PyG self-loops.
// ---------------------------------------------------------------------------
__global__ void k_edge_max(const int* __restrict__ ei, const float* __restrict__ as_,
                           const float* __restrict__ ad_, unsigned* __restrict__ emax,
                           int H_) {
  int t = blockIdx.x * blockDim.x + threadIdx.x;
  int tot = (N_EDGES + N_NODES) * H_;
  if (t >= tot) return;
  int e = t / H_, hd = t % H_;
  int s, d;
  if (e < N_EDGES) { s = ei[e]; d = ei[N_EDGES + e]; } else { s = d = e - N_EDGES; }
  float v = as_[s * H_ + hd] + ad_[d * H_ + hd];
  v = (v >= 0.f) ? v : NEG_SLOPE * v;
  atomicMax(&emax[d * H_ + hd], f2ord(v));
}

// edge pass 2: w = exp(e - max); denom[dst] += w
__global__ void k_edge_w(const int* __restrict__ ei, const float* __restrict__ as_,
                         const float* __restrict__ ad_, const unsigned* __restrict__ emax,
                         float* __restrict__ den, float* __restrict__ wbuf, int H_) {
  int t = blockIdx.x * blockDim.x + threadIdx.x;
  int tot = (N_EDGES + N_NODES) * H_;
  if (t >= tot) return;
  int e = t / H_, hd = t % H_;
  int s, d;
  if (e < N_EDGES) { s = ei[e]; d = ei[N_EDGES + e]; } else { s = d = e - N_EDGES; }
  float v = as_[s * H_ + hd] + ad_[d * H_ + hd];
  v = (v >= 0.f) ? v : NEG_SLOPE * v;
  float w = __expf(v - ord2f(emax[d * H_ + hd]));
  wbuf[t] = w;
  atomicAdd(&den[d * H_ + hd], w);
}

// edge pass 3: one wave per edge; agg[dst] += h[src] * alpha (coalesced, f32 atomics)
__global__ void k_msg(const int* __restrict__ ei, const float* __restrict__ wbuf,
                      const float* __restrict__ den, const float* __restrict__ h,
                      float* __restrict__ agg, int H_, int F_) {
  int wv = (blockIdx.x * blockDim.x + threadIdx.x) >> 5;  // wave id = edge id
  int lane = threadIdx.x & 31;
  if (wv >= N_EDGES + N_NODES) return;
  int s, d;
  if (wv < N_EDGES) { s = ei[wv]; d = ei[N_EDGES + wv]; } else { s = d = wv - N_EDGES; }
  float aown = 0.f;
  if (lane < H_) aown = wbuf[wv * H_ + lane] / den[d * H_ + lane];
  const float* hs = h + (size_t)s * F_;
  float* ag = agg + (size_t)d * F_;
  for (int c = lane; c < F_; c += 32) {
    int head = c >> 7;                       // c / CDIM
    float alpha = __shfl(aown, head, 32);    // wave32 broadcast of per-head alpha
    atomicAdd(&ag[c], hs[c] * alpha);
  }
}

// bias + eval BatchNorm (+optional ReLU); writes fp32 in place and bf16 copy
__global__ void k_post(float* __restrict__ agg, __bf16* __restrict__ hbf,
                       const float* __restrict__ bias, const float* __restrict__ g,
                       const float* __restrict__ be, int F_, int relu) {
  long t = (long)blockIdx.x * blockDim.x + threadIdx.x;
  if (t >= (long)N_NODES * F_) return;
  int c = (int)(t % F_);
  float inv = rsqrtf(1.0f + 1e-5f);
  float v = (agg[t] + bias[c]) * (g[c] * inv) + be[c];
  if (relu) v = fmaxf(v, 0.f);
  agg[t] = v;
  hbf[t] = (__bf16)v;
}

// classifier hidden: hid[n][j] = relu(x[n,:128] . Wc1[:,j] + bc1[j])
__global__ void k_cls_hidden(const float* __restrict__ x, const float* __restrict__ W,
                             const float* __restrict__ b, float* __restrict__ hid) {
  int t = blockIdx.x * blockDim.x + threadIdx.x;
  if (t >= N_NODES * 64) return;
  int n = t >> 6, j = t & 63;
  const float* xr = x + (size_t)n * CDIM;
  float acc = b[j];
#pragma unroll 8
  for (int k = 0; k < CDIM; ++k) acc += xr[k] * W[k * 64 + j];
  hid[t] = fmaxf(acc, 0.f);
}

// classifier logits + log_softmax: one wave32 per node, lane handles class j and j+32
__global__ void k_cls_out(const float* __restrict__ hid, const float* __restrict__ W,
                          const float* __restrict__ b, float* __restrict__ out) {
  int g = blockIdx.x * blockDim.x + threadIdx.x;
  int n = g >> 5, lane = g & 31;
  if (n >= N_NODES) return;
  const float* hr = hid + (size_t)n * 64;
  float l0 = b[lane];
  float l1 = (lane < 8) ? b[lane + 32] : -1e30f;
#pragma unroll 8
  for (int k = 0; k < 64; ++k) {
    float hv = hr[k];
    l0 += hv * W[k * NCLS + lane];
    if (lane < 8) l1 += hv * W[k * NCLS + lane + 32];
  }
  float m = fmaxf(l0, l1);
  for (int off = 16; off; off >>= 1) m = fmaxf(m, __shfl_xor(m, off, 32));
  float se = __expf(l0 - m) + ((lane < 8) ? __expf(l1 - m) : 0.f);
  for (int off = 16; off; off >>= 1) se += __shfl_xor(se, off, 32);
  float lse = m + __logf(se);
  out[(size_t)n * NCLS + lane] = l0 - lse;
  if (lane < 8) out[(size_t)n * NCLS + lane + 32] = l1 - lse;
}

// ---------------------------------------------------------------------------
extern "C" void kernel_launch(void* const* d_in, const int* in_sizes, int n_in,
                              void* d_out, int out_size, void* d_ws, size_t ws_size,
                              hipStream_t stream) {
  (void)in_sizes; (void)n_in; (void)out_size; (void)ws_size;
  const float* x   = (const float*)d_in[0];
  const int*   ei  = (const int*)d_in[1];
  const float* W1  = (const float*)d_in[2];
  const float* a1s = (const float*)d_in[3];
  const float* a1d = (const float*)d_in[4];
  const float* b1  = (const float*)d_in[5];
  const float* g1  = (const float*)d_in[6];
  const float* be1 = (const float*)d_in[7];
  const float* W2  = (const float*)d_in[8];
  const float* a2s = (const float*)d_in[9];
  const float* a2d = (const float*)d_in[10];
  const float* b2  = (const float*)d_in[11];
  const float* g2  = (const float*)d_in[12];
  const float* be2 = (const float*)d_in[13];
  const float* W3  = (const float*)d_in[14];
  const float* a3s = (const float*)d_in[15];
  const float* a3d = (const float*)d_in[16];
  const float* b3  = (const float*)d_in[17];
  const float* g3  = (const float*)d_in[18];
  const float* be3 = (const float*)d_in[19];
  const float* Wc1 = (const float*)d_in[20];
  const float* bc1 = (const float*)d_in[21];
  const float* Wc2 = (const float*)d_in[22];
  const float* bc2 = (const float*)d_in[23];
  float* out = (float*)d_out;

  // workspace carve-up (256B aligned slabs)
  char* ws = (char*)d_ws;
  size_t off = 0;
  auto carve = [&](size_t bytes) -> void* {
    void* p = ws + off;
    off += (bytes + 255) & ~(size_t)255;
    return p;
  };
  float*    hbuf = (float*)carve((size_t)N_NODES * FDIM * 4);   // transformed features
  float*    agg  = (float*)carve((size_t)N_NODES * FDIM * 4);   // aggregation / layer output
  __bf16*   hbf  = (__bf16*)carve((size_t)N_NODES * FDIM * 2);  // bf16 GEMM operand
  __bf16*   wt   = (__bf16*)carve((size_t)FDIM * FDIM * 2);     // transposed bf16 weights
  float*    asb  = (float*)carve((size_t)N_NODES * HEADS * 4);
  float*    adb  = (float*)carve((size_t)N_NODES * HEADS * 4);
  unsigned* emax = (unsigned*)carve((size_t)N_NODES * HEADS * 4);
  float*    den  = (float*)carve((size_t)N_NODES * HEADS * 4);
  float*    wbuf = (float*)carve((size_t)(N_EDGES + N_NODES) * HEADS * 4);
  float*    hid  = (float*)carve((size_t)N_NODES * 64 * 4);

  auto run_layer = [&](int K, int Nout, const float* W, const float* asw,
                       const float* adw, const float* bias, const float* gam,
                       const float* bet, int H_, int relu) {
    int wn = K * Nout;
    k_wtrans<<<(wn + 255) / 256, 256, 0, stream>>>(W, wt, K, Nout);
    dim3 gg((N_NODES + 127) / 128, Nout / 64);
    k_gemm_bf16<<<gg, 256, 0, stream>>>(hbf, wt, hbuf, N_NODES, K, Nout);
    int nh = N_NODES * H_;
    k_alpha<<<(nh + 255) / 256, 256, 0, stream>>>(hbuf, asw, adw, asb, adb, H_, Nout);
    hipMemsetAsync(emax, 0, (size_t)nh * 4, stream);
    hipMemsetAsync(den, 0, (size_t)nh * 4, stream);
    hipMemsetAsync(agg, 0, (size_t)N_NODES * Nout * 4, stream);
    long teh = (long)(N_EDGES + N_NODES) * H_;
    k_edge_max<<<(int)((teh + 255) / 256), 256, 0, stream>>>(ei, asb, adb, emax, H_);
    k_edge_w<<<(int)((teh + 255) / 256), 256, 0, stream>>>(ei, asb, adb, emax, den, wbuf, H_);
    long mt = (long)(N_EDGES + N_NODES) * 32;
    k_msg<<<(int)((mt + 255) / 256), 256, 0, stream>>>(ei, wbuf, den, hbuf, agg, H_, Nout);
    long pt = (long)N_NODES * Nout;
    k_post<<<(int)((pt + 255) / 256), 256, 0, stream>>>(agg, hbf, bias, gam, bet, Nout, relu);
  };

  // layer 1 input: cast x -> bf16 operand buffer
  k_cast_bf16<<<2048, 256, 0, stream>>>(x, hbf, (long)N_NODES * F_IN);
  run_layer(F_IN, FDIM, W1, a1s, a1d, b1, g1, be1, HEADS, 1);
  run_layer(FDIM, FDIM, W2, a2s, a2d, b2, g2, be2, HEADS, 1);
  run_layer(FDIM, CDIM, W3, a3s, a3d, b3, g3, be3, 1, 0);

  // classifier on agg [N,128] (post-BN output, fp32)
  k_cls_hidden<<<(N_NODES * 64 + 255) / 256, 256, 0, stream>>>(agg, Wc1, bc1, hid);
  long ct = (long)N_NODES * 32;
  k_cls_out<<<(int)((ct + 255) / 256), 256, 0, stream>>>(hid, Wc2, bc2, out);
}
